// MLPPrediction_60026462929563
// MI455X (gfx1250) — compile-verified, matrix operand order checked by reading, not run on hardware
//
#include <hip/hip_runtime.h>

typedef __attribute__((ext_vector_type(16))) _Float16 v16h;
typedef __attribute__((ext_vector_type(8)))  _Float16 v8h;
typedef __attribute__((ext_vector_type(8)))  float    v8f;

#define IN_FEAT   128
#define K_TOTAL   256            // 2 * IN_FEAT (concat of src/dst features)
#define LDS_KPAD  264            // padded row stride (halves) to avoid bank conflicts
#define WAVES_PER_BLOCK 8
#define THREADS   (WAVES_PER_BLOCK * 32)

// Convert 16 f32 feature values (two 8-wide chunks) to an f16 A-fragment
// matching the 16-bit 16x32 A layout (lane half h picks K sub-chunks).
__device__ __forceinline__ v16h cvt_a(const float* __restrict__ row, int k0) {
    float4 x0 = *(const float4*)(row + k0);
    float4 x1 = *(const float4*)(row + k0 + 4);
    float4 x2 = *(const float4*)(row + k0 + 16);
    float4 x3 = *(const float4*)(row + k0 + 20);
    v16h a;
    a[0]  = (_Float16)x0.x; a[1]  = (_Float16)x0.y;
    a[2]  = (_Float16)x0.z; a[3]  = (_Float16)x0.w;
    a[4]  = (_Float16)x1.x; a[5]  = (_Float16)x1.y;
    a[6]  = (_Float16)x1.z; a[7]  = (_Float16)x1.w;
    a[8]  = (_Float16)x2.x; a[9]  = (_Float16)x2.y;
    a[10] = (_Float16)x2.z; a[11] = (_Float16)x2.w;
    a[12] = (_Float16)x3.x; a[13] = (_Float16)x3.y;
    a[14] = (_Float16)x3.z; a[15] = (_Float16)x3.w;
    return a;
}

// Layer-2 epilogue for one 16-edge sub-tile: bias + ReLU + dot(W2) in registers,
// then a 4-step shuffle butterfly across each 16-lane half.
__device__ __forceinline__ void epilogue(const v8f* __restrict__ acc,
                                         const float* __restrict__ b1v,
                                         const float* __restrict__ w2v,
                                         float b2v, int eBase, int h, int l,
                                         int E, float* __restrict__ out) {
    float s[8];
#pragma unroll
    for (int r = 0; r < 8; ++r) s[r] = 0.0f;
#pragma unroll
    for (int nb = 0; nb < 8; ++nb) {
#pragma unroll
        for (int r = 0; r < 8; ++r) {
            float hv = acc[nb][r] + b1v[nb];
            hv = hv > 0.0f ? hv : 0.0f;
            s[r] += hv * w2v[nb];
        }
    }
#pragma unroll
    for (int mask = 1; mask <= 8; mask <<= 1) {
#pragma unroll
        for (int r = 0; r < 8; ++r)
            s[r] += __shfl_xor(s[r], mask, 32);
    }
    if (l == 0) {   // lane 0 -> edges 0..7, lane 16 -> edges 8..15
        const int base = eBase + h * 8;
        if (base + 8 <= E) {
            float4 o0 = make_float4(s[0] + b2v, s[1] + b2v, s[2] + b2v, s[3] + b2v);
            float4 o1 = make_float4(s[4] + b2v, s[5] + b2v, s[6] + b2v, s[7] + b2v);
            *(float4*)(out + base)     = o0;
            *(float4*)(out + base + 4) = o1;
        } else {
#pragma unroll
            for (int r = 0; r < 8; ++r)
                if (base + r < E) out[base + r] = s[r] + b2v;
        }
    }
}

__global__ __launch_bounds__(THREADS)
void edge_mlp_wmma_kernel(const float* __restrict__ feature,
                          const long long* __restrict__ src,
                          const long long* __restrict__ dst,
                          const float* __restrict__ W1,
                          const float* __restrict__ b1,
                          const float* __restrict__ W2,
                          const float* __restrict__ b2,
                          float* __restrict__ out,
                          int E, int nTiles2)
{
    extern __shared__ __align__(16) unsigned char smem_raw[];
    _Float16* __restrict__ W1h = (_Float16*)smem_raw;   // [128][LDS_KPAD] f16

    const int tid  = threadIdx.x;
    const int wave = tid >> 5;
    const int lane = tid & 31;
    const int h    = lane >> 4;   // lane half selects K sub-chunk
    const int l    = lane & 15;   // row (A: edge) / col (B: output) in tile

    // ---- stage W1 (f32 -> f16) into LDS, cooperative across block ----
    for (int i = tid; i < IN_FEAT * (K_TOTAL / 4); i += THREADS) {
        int n = i >> 6;          // output row (64 float4 per row)
        int c = i & 63;          // float4 index within row
        float4 v = ((const float4*)(W1 + n * K_TOTAL))[c];
        _Float16* p = W1h + n * LDS_KPAD + c * 4;
        p[0] = (_Float16)v.x; p[1] = (_Float16)v.y;
        p[2] = (_Float16)v.z; p[3] = (_Float16)v.w;
    }
    __syncthreads();

    // ---- per-lane layer-2 constants ----
    float b1v[8], w2v[8];
#pragma unroll
    for (int nb = 0; nb < 8; ++nb) {
        b1v[nb] = b1[nb * 16 + l];
        w2v[nb] = W2[nb * 16 + l];
    }
    const float b2v = b2[0];

    const int tStride = gridDim.x * WAVES_PER_BLOCK;

    // ---- grid-stride loop over 32-edge macro-tiles (2 x 16-edge WMMA tiles) ----
    for (int t = blockIdx.x * WAVES_PER_BLOCK + wave; t < nTiles2; t += tStride) {
        const int eBase = t * 32;
        int e0 = eBase + l;      if (e0 >= E) e0 = E - 1;
        int e1 = eBase + 16 + l; if (e1 >= E) e1 = E - 1;
        const float* srow0 = feature + (size_t)src[e0] * IN_FEAT;
        const float* drow0 = feature + (size_t)dst[e0] * IN_FEAT;
        const float* srow1 = feature + (size_t)src[e1] * IN_FEAT;
        const float* drow1 = feature + (size_t)dst[e1] * IN_FEAT;

        // Prefetch next macro-tile's feature rows (per-lane addresses: one
        // global_prefetch_b8 covers 16 distinct gathered cache lines).
        const int tn = t + tStride;
        if (tn < nTiles2) {
            int en0 = tn * 32 + l;      if (en0 >= E) en0 = E - 1;
            int en1 = tn * 32 + 16 + l; if (en1 >= E) en1 = E - 1;
            const char* ps0 = (const char*)(feature + (size_t)src[en0] * IN_FEAT);
            const char* pd0 = (const char*)(feature + (size_t)dst[en0] * IN_FEAT);
            const char* ps1 = (const char*)(feature + (size_t)src[en1] * IN_FEAT);
            const char* pd1 = (const char*)(feature + (size_t)dst[en1] * IN_FEAT);
#pragma unroll
            for (int off = 0; off < 512; off += 128) {   // 512 B per feature row
                __builtin_prefetch(ps0 + off, 0, 1);
                __builtin_prefetch(pd0 + off, 0, 1);
                __builtin_prefetch(ps1 + off, 0, 1);
                __builtin_prefetch(pd1 + off, 0, 1);
            }
        }

        // Opaque zero redefined every iteration: B addresses become loop-variant,
        // so the compiler cannot hoist (and spill) the LDS B-fragment loads.
        unsigned ldsOff;
        asm volatile("s_mov_b32 %0, 0" : "=s"(ldsOff));
        const _Float16* W1base = (const _Float16*)((const char*)W1h + ldsOff);

        v8f acc0[8] = {};   // sub-tile 0: edges eBase .. eBase+15
        v8f acc1[8] = {};   // sub-tile 1: edges eBase+16 .. eBase+31

#pragma unroll
        for (int kb = 0; kb < 8; ++kb) {
            // K blocks 0..3 read src features, 4..7 read dst features
            const int k0 = ((kb & 3) << 5) + (h << 3);
            v16h a0 = cvt_a((kb < 4) ? srow0 : drow0, k0);
            v16h a1 = cvt_a((kb < 4) ? srow1 : drow1, k0);

            const _Float16* bbase = W1base + kb * 32 + h * 8;
            auto loadB = [&](int nb) -> v16h {
                const _Float16* bp = bbase + (size_t)(nb * 16 + l) * LDS_KPAD;
                v8h blo = *(const v8h*)bp;          // K = kb*32 + h*8 .. +7
                v8h bhi = *(const v8h*)(bp + 16);   // K = kb*32 + 16 + h*8 ..
                v16h bf;
#pragma unroll
                for (int i = 0; i < 8; ++i) { bf[i] = blo[i]; bf[8 + i] = bhi[i]; }
                return bf;
            };

            // Software-pipelined B: issue nb+1's LDS loads before nb's WMMAs so
            // the wait becomes s_wait_dscnt 2 instead of 0 (latency overlapped).
            v16h bcur = loadB(0);
#pragma unroll
            for (int nb = 0; nb < 8; ++nb) {
                v16h bnext = bcur;
                if (nb < 7) bnext = loadB(nb + 1);
                // one B fragment feeds two independent WMMAs (no RAW hazard)
                acc0[nb] = __builtin_amdgcn_wmma_f32_16x16x32_f16(
                    false, a0, false, bcur, (short)0, acc0[nb], false, false);
                acc1[nb] = __builtin_amdgcn_wmma_f32_16x16x32_f16(
                    false, a1, false, bcur, (short)0, acc1[nb], false, false);
                bcur = bnext;
            }
        }

        epilogue(acc0, b1v, w2v, b2v, eBase,      h, l, E, out);
        epilogue(acc1, b1v, w2v, b2v, eBase + 16, h, l, E, out);
    }
}

extern "C" void kernel_launch(void* const* d_in, const int* in_sizes, int n_in,
                              void* d_out, int out_size, void* d_ws, size_t ws_size,
                              hipStream_t stream) {
    (void)n_in; (void)out_size; (void)d_ws; (void)ws_size;
    const float*     feature = (const float*)d_in[0];
    const long long* src     = (const long long*)d_in[1];
    const long long* dst     = (const long long*)d_in[2];
    const float*     W1      = (const float*)d_in[3];
    const float*     b1      = (const float*)d_in[4];
    const float*     W2      = (const float*)d_in[5];
    const float*     b2      = (const float*)d_in[6];
    float*           out     = (float*)d_out;

    const int E       = in_sizes[1];
    const int nTiles2 = (E + 31) / 32;
    int blocks = (nTiles2 + WAVES_PER_BLOCK - 1) / WAVES_PER_BLOCK;
    if (blocks > 768) blocks = 768;          // persistent blocks: amortize W1 staging
    const size_t lds = (size_t)IN_FEAT * LDS_KPAD * 2;  // f16 W1 in LDS (~66 KB)

    edge_mlp_wmma_kernel<<<blocks, THREADS, lds, stream>>>(
        feature, src, dst, W1, b1, W2, b2, out, E, nTiles2);
}
